// VGG_EDAttention_49993419325946
// MI455X (gfx1250) — matrix operand order repaired and checked by reading, not compiled, not run on hardware
//
#include <hip/hip_runtime.h>
#include <hip/hip_bf16.h>
#include <math.h>

typedef __attribute__((ext_vector_type(16))) _Float16 v16h;
typedef __attribute__((ext_vector_type(8)))  _Float16 v8h;
typedef __attribute__((ext_vector_type(8)))  float    v8f;

#define BN_EPS 1e-5f
#define L_SA   3136   // 56*56, shared attention sequence length

// ---------------------------------------------------------------------------
// Average pool, window=stride=k
// ---------------------------------------------------------------------------
__global__ void avgpool_k(const float* __restrict__ in, float* __restrict__ out,
                          int C, int Hin, int k) {
  int Hout = Hin / k;
  int total = C * Hout * Hout;
  int i = blockIdx.x * blockDim.x + threadIdx.x;
  if (i >= total) return;
  int c = i / (Hout * Hout), r = i % (Hout * Hout);
  int y = r / Hout, x = r % Hout;
  const float* base = in + c * Hin * Hin;
  float s = 0.f;
  for (int dy = 0; dy < k; ++dy)
    for (int dx = 0; dx < k; ++dx)
      s += base[(y * k + dy) * Hin + (x * k + dx)];
  out[i] = s / (float)(k * k);
}

// ---------------------------------------------------------------------------
// 2x2 max pool
// ---------------------------------------------------------------------------
__global__ void maxpool2_k(const float* __restrict__ in, float* __restrict__ out,
                           int C, int H) {
  int Ho = H / 2;
  int total = C * Ho * Ho;
  int i = blockIdx.x * blockDim.x + threadIdx.x;
  if (i >= total) return;
  int c = i / (Ho * Ho), r = i % (Ho * Ho);
  int y = r / Ho, x = r % Ho;
  const float* base = in + c * H * H + (2 * y) * H + 2 * x;
  out[i] = fmaxf(fmaxf(base[0], base[1]), fmaxf(base[H], base[H + 1]));
}

// ---------------------------------------------------------------------------
// q/k/v = Linear(1,1) elementwise
// ---------------------------------------------------------------------------
__global__ void qkv_k(const float* __restrict__ sa,
                      float* __restrict__ q, float* __restrict__ k, float* __restrict__ v,
                      const float* wq, const float* bq,
                      const float* wk, const float* bk,
                      const float* wv, const float* bv, int n) {
  int i = blockIdx.x * blockDim.x + threadIdx.x;
  if (i >= n) return;
  float s = sa[i];
  q[i] = wq[0] * s + bq[0];
  k[i] = wk[0] * s + bk[0];
  v[i] = wv[0] * s + bv[0];
}

// ---------------------------------------------------------------------------
// attn_equal: out[c,i] = sum_j softmax_j(q[c,i]*k[c,j]) * v[c,j]
// ---------------------------------------------------------------------------
__global__ void attn_equal_k(const float* __restrict__ q,
                             const float* __restrict__ k,
                             const float* __restrict__ v,
                             float* __restrict__ out) {
  __shared__ float sk[L_SA];
  __shared__ float sv[L_SA];
  __shared__ float smax[256];
  __shared__ float smin[256];
  int tid = threadIdx.x;
  int c = blockIdx.y;
  const float* kc = k + c * L_SA;
  const float* vc = v + c * L_SA;
  float mx = -1e30f, mn = 1e30f;
  for (int j = tid; j < L_SA; j += blockDim.x) {
    float kj = kc[j];
    sk[j] = kj;
    sv[j] = vc[j];
    mx = fmaxf(mx, kj);
    mn = fminf(mn, kj);
  }
  smax[tid] = mx; smin[tid] = mn;
  __syncthreads();
  for (int s = 128; s > 0; s >>= 1) {
    if (tid < s) {
      smax[tid] = fmaxf(smax[tid], smax[tid + s]);
      smin[tid] = fminf(smin[tid], smin[tid + s]);
    }
    __syncthreads();
  }
  float kmax = smax[0], kmin = smin[0];
  int i = blockIdx.x * blockDim.x + tid;
  if (i >= L_SA) return;
  float t = q[c * L_SA + i];
  float m = (t >= 0.f) ? t * kmax : t * kmin;
  float se = 0.f, sw = 0.f;
  for (int j = 0; j < L_SA; ++j) {
    float e = __expf(t * sk[j] - m);
    se += e;
    sw += e * sv[j];
  }
  out[c * L_SA + i] = sw / se;
}

// ---------------------------------------------------------------------------
// km = mean_c k, vm = mean_c v_att ; scal[0]=max(km), scal[1]=min(km)
// ---------------------------------------------------------------------------
__global__ void kmvm_k(const float* __restrict__ k, const float* __restrict__ vatt,
                       float* __restrict__ km, float* __restrict__ vm,
                       float* __restrict__ scal) {
  __shared__ float smax[256];
  __shared__ float smin[256];
  int tid = threadIdx.x;
  float mx = -1e30f, mn = 1e30f;
  for (int j = tid; j < L_SA; j += blockDim.x) {
    float kj = (k[j] + k[L_SA + j] + k[2 * L_SA + j]) * (1.f / 3.f);
    float vj = (vatt[j] + vatt[L_SA + j] + vatt[2 * L_SA + j]) * (1.f / 3.f);
    km[j] = kj;
    vm[j] = vj;
    mx = fmaxf(mx, kj);
    mn = fminf(mn, kj);
  }
  smax[tid] = mx; smin[tid] = mn;
  __syncthreads();
  for (int s = 128; s > 0; s >>= 1) {
    if (tid < s) {
      smax[tid] = fmaxf(smax[tid], smax[tid + s]);
      smin[tid] = fminf(smin[tid], smin[tid + s]);
    }
    __syncthreads();
  }
  if (tid == 0) { scal[0] = smax[0]; scal[1] = smin[0]; }
}

// ---------------------------------------------------------------------------
// attn_asym: t = lin(conv_x[row]); out[row] = sum_j exp(t*km[j]-m)*vm[j] / Z
// ---------------------------------------------------------------------------
__global__ void attn_asym_k(const float* __restrict__ in, float* __restrict__ out,
                            const float* __restrict__ km, const float* __restrict__ vm,
                            const float* __restrict__ scal,
                            const float* lw, const float* lb, int N) {
  __shared__ float skm[L_SA];
  __shared__ float svm[L_SA];
  for (int j = threadIdx.x; j < L_SA; j += blockDim.x) {
    skm[j] = km[j];
    svm[j] = vm[j];
  }
  __syncthreads();
  int i = blockIdx.x * blockDim.x + threadIdx.x;
  if (i >= N) return;
  float kmax = scal[0], kmin = scal[1];
  float t = lw[0] * in[i] + lb[0];
  float m = (t >= 0.f) ? t * kmax : t * kmin;
  float se = 0.f, sw = 0.f;
  for (int j = 0; j < L_SA; ++j) {
    float e = __expf(t * skm[j] - m);
    se += e;
    sw += e * svm[j];
  }
  out[i] = sw / se;
}

// ---------------------------------------------------------------------------
// Weight pre-conversion: f32 [Co][Ktot] -> f16 [Co][Kp], zero-padded K
// ---------------------------------------------------------------------------
__global__ void wcvt_k(const float* __restrict__ w, _Float16* __restrict__ out,
                       int Co, int Ktot, int Kp) {
  int i = blockIdx.x * blockDim.x + threadIdx.x;
  if (i >= Co * Kp) return;
  int m = i / Kp, kk = i - m * Kp;
  out[i] = (kk < Ktot) ? (_Float16)w[m * Ktot + kk] : (_Float16)0.f;
}

// ---------------------------------------------------------------------------
// 3x3 conv (pad 1) + bias + BN + ReLU, implicit-GEMM on WMMA f16->f32.
// Block = 8 waves sharing a 32-column N tile (two 16-col subtiles); each wave
// owns one 16-row M tile and issues TWO wmma per K-step, reusing its A
// fragment for both subtiles (2x arithmetic intensity per weight load).
// B tiles (2 x 32x16 f16) staged in double-buffered LDS by all 256 threads
// (branchless clamped gathers -> packed ds_store_b64); consumed as
// ds_load_b128 pairs. A fragment: two unconditional global 16B loads from
// pre-converted, K-padded f16 weights.
// Fragment layouts (wave32):
//   A 16x32 f16: lane(half h, l16): v[i] = A[l16][16h + i]
//   B 32x16 f16: lane: v[i] = B[16h + i][l16]
//   D 16x16 f32: v[r] -> row 8h + r, col l16
// ---------------------------------------------------------------------------
__global__ __launch_bounds__(256)
void conv3x3_wmma_k(const float* __restrict__ in, const _Float16* __restrict__ w16,
                    const float* __restrict__ bias, const float* __restrict__ gam,
                    const float* __restrict__ bet, const float* __restrict__ rmean,
                    const float* __restrict__ rvar, float* __restrict__ out,
                    int Ci, int Co, int H, int Kp) {
  const int W = H;
  const int HW = H * W;
  const int Ktot = Ci * 9;
  __shared__ _Float16 sB[2][2][16 * 32];   // [buf][subtile][col16*32 + k]

  const int tid = threadIdx.x;
  const int lane = tid & 31;
  const int wid = tid >> 5;
  const int half = lane >> 4;
  const int l16 = lane & 15;
  const int n_base = blockIdx.x * 32;
  const int co_base = blockIdx.y * 128 + wid * 16;
  const bool mvalid = (co_base < Co);        // wave-uniform

  // Staging role: column sc (0..31), k-quad kg (0..7) -> 4 elems per thread
  const int sc = tid & 31;
  const int kg = tid >> 5;
  const int n_s = n_base + sc;
  const bool nsv = (n_s < HW);
  const int ys = nsv ? (n_s / W) : 0;
  const int xs = nsv ? (n_s % W) : 0;
  const int s_sub = sc >> 4;                 // subtile of staged column
  const int s_c16 = sc & 15;

  // Consumer indices (two columns per lane)
  const int n0 = n_base + l16;
  const int n1 = n_base + 16 + l16;
  const _Float16* wrow = w16 + (size_t)(mvalid ? co_base + l16 : 0) * (size_t)Kp;

  auto stage = [&](int buf, int kb) {
    float vals[4];
#pragma unroll
    for (int t = 0; t < 4; ++t) {
      int kk = kb + 4 * kg + t;
      int ci = kk / 9;
      int r = kk - ci * 9;
      int iy = ys + (r / 3) - 1;
      int ix = xs + (r - (r / 3) * 3) - 1;
      bool ok = nsv && (kk < Ktot) && (iy >= 0) && (iy < H) && (ix >= 0) && (ix < W);
      int addr = ok ? (ci * HW + iy * W + ix) : 0;
      float v = in[addr];                    // unconditional load, clamped addr
      vals[t] = ok ? v : 0.f;
    }
    union { _Float16 h[4]; unsigned long long u; } p;
#pragma unroll
    for (int t = 0; t < 4; ++t) p.h[t] = (_Float16)vals[t];
    // one ds_store_b64 at f16 offset s_c16*32 + 4*kg (8B aligned)
    ((unsigned long long*)sB[buf][s_sub])[s_c16 * 8 + kg] = p.u;
  };

  v8f acc0 = {}, acc1 = {};
  const int nk = (Ktot + 31) >> 5;
  stage(0, 0);
  for (int ik = 0; ik < nk; ++ik) {
    __syncthreads();
    if (ik + 1 < nk) stage((ik + 1) & 1, (ik + 1) << 5);
    if (mvalid) {
      const int kb = ik << 5;
      v8h alo = *(const v8h*)(wrow + kb + 16 * half);       // K = kb+16h .. +7
      v8h ahi = *(const v8h*)(wrow + kb + 16 * half + 8);   // K = kb+16h+8 .. +7
      v16h a;
#pragma unroll
      for (int i = 0; i < 8; ++i) { a[i] = alo[i]; a[8 + i] = ahi[i]; }
      v16h b0 = *(const v16h*)(sB[ik & 1][0] + l16 * 32 + 16 * half);
      v16h b1 = *(const v16h*)(sB[ik & 1][1] + l16 * 32 + 16 * half);
      acc0 = __builtin_amdgcn_wmma_f32_16x16x32_f16(
          false, a, false, b0, (short)0, acc0, false, false);
      acc1 = __builtin_amdgcn_wmma_f32_16x16x32_f16(
          false, a, false, b1, (short)0, acc1, false, false);
    }
  }

  if (mvalid) {
#pragma unroll
    for (int r = 0; r < 8; ++r) {
      int mm = co_base + (half << 3) + r;
      float sc_ = gam[mm] / sqrtf(rvar[mm] + BN_EPS);
      float sh_ = (bias[mm] - rmean[mm]) * sc_ + bet[mm];
      if (n0 < HW) out[mm * HW + n0] = fmaxf(acc0[r] * sc_ + sh_, 0.f);
      if (n1 < HW) out[mm * HW + n1] = fmaxf(acc1[r] * sc_ + sh_, 0.f);
    }
  }
}

// ---------------------------------------------------------------------------
// GEMV + bias + BN (+optional ReLU). One wave per row; float4 loads + weight
// prefetch. fc1 streams 565MB of weights -> HBM bound (~24us @ 23.3TB/s).
// ---------------------------------------------------------------------------
__global__ __launch_bounds__(256)
void gemv_bn_k(const float* __restrict__ Wm, const float* __restrict__ x,
               const float* __restrict__ bias,
               const float* __restrict__ bg, const float* __restrict__ bb,
               const float* __restrict__ brm, const float* __restrict__ brv,
               float* __restrict__ out, int In, int Out, int relu) {
  int wid = threadIdx.x >> 5;
  int lane = threadIdx.x & 31;
  int row = blockIdx.x * 8 + wid;
  if (row >= Out) return;
  const float* Wr = Wm + (size_t)row * (size_t)In;
  float s = 0.f;
  for (int i = lane * 4; i < In; i += 32 * 4) {
    __builtin_prefetch(Wr + i + 1024, 0, 1);   // global_prefetch_b8, speculative
    float4 wv = *(const float4*)(Wr + i);
    float4 xv = *(const float4*)(x + i);
    s += wv.x * xv.x + wv.y * xv.y + wv.z * xv.z + wv.w * xv.w;
  }
  for (int off = 16; off > 0; off >>= 1) s += __shfl_down(s, off, 32);
  if (lane == 0) {
    float y = s + bias[row];
    y = (y - brm[row]) * (bg[row] / sqrtf(brv[row] + BN_EPS)) + bb[row];
    if (relu) y = fmaxf(y, 0.f);
    out[row] = y;
  }
}

// ---------------------------------------------------------------------------
// Output layer (10x4096) + softmax, single block of 10 waves
// ---------------------------------------------------------------------------
__global__ __launch_bounds__(320)
void out_softmax_k(const float* __restrict__ Wo, const float* __restrict__ bo,
                   const float* __restrict__ h, float* __restrict__ out) {
  __shared__ float logits[10];
  int wid = threadIdx.x >> 5;
  int lane = threadIdx.x & 31;
  if (wid < 10) {
    const float* Wr = Wo + wid * 4096;
    float s = 0.f;
    for (int i = lane; i < 4096; i += 32) s += Wr[i] * h[i];
    for (int off = 16; off > 0; off >>= 1) s += __shfl_down(s, off, 32);
    if (lane == 0) logits[wid] = s + bo[wid];
  }
  __syncthreads();
  if (threadIdx.x == 0) {
    float m = -1e30f;
    for (int i = 0; i < 10; ++i) m = fmaxf(m, logits[i]);
    float e[10], se = 0.f;
    for (int i = 0; i < 10; ++i) { e[i] = __expf(logits[i] - m); se += e[i]; }
    for (int i = 0; i < 10; ++i) out[i] = e[i] / se;
  }
}

// ---------------------------------------------------------------------------
// final_x = concat(v_att[9408], conv_x[25088]); write ws copy (aligned) + d_out
// ---------------------------------------------------------------------------
__global__ void concat_k(const float* __restrict__ vatt, const float* __restrict__ convx,
                         float* __restrict__ fx, float* __restrict__ dout) {
  int i = blockIdx.x * blockDim.x + threadIdx.x;
  if (i >= 34496) return;
  float val = (i < 9408) ? vatt[i] : convx[i - 9408];
  fx[i] = val;
  dout[i] = val;
}

// ---------------------------------------------------------------------------
// Host launch
// ---------------------------------------------------------------------------
extern "C" void kernel_launch(void* const* d_in, const int* in_sizes, int n_in,
                              void* d_out, int out_size, void* d_ws, size_t ws_size,
                              hipStream_t stream) {
  // Semantic slots (dict-insertion order of setup_inputs):
  // 0:x ; 1..60: 10 convs x (w,b,g,bt,rm,rv) ; 61,62:query ; 63,64:key ;
  // 65,66:value ; 67..74: att_lin x4 ; 75,76:fc1 ; 77..80:bn1 ; 81,82:fc2 ;
  // 83..86:bn2 ; 87,88:out
  const float* in[89];
  if (in_sizes[0] == 150528) {
    for (int i = 0; i < 89 && i < n_in; ++i) in[i] = (const float*)d_in[i];
  } else {
    // jax pytree (sorted-key) order: att_lin, blocks, bn1, bn2, fc1, fc2,
    // key, out, query, value, then x last.
    in[0] = (const float*)d_in[88];
    for (int i = 0; i < 60; ++i) in[1 + i] = (const float*)d_in[8 + i];
    in[61] = (const float*)d_in[84]; in[62] = (const float*)d_in[85];    // query
    in[63] = (const float*)d_in[80]; in[64] = (const float*)d_in[81];    // key
    in[65] = (const float*)d_in[86]; in[66] = (const float*)d_in[87];    // value
    for (int i = 0; i < 8; ++i) in[67 + i] = (const float*)d_in[i];      // att_lin
    in[75] = (const float*)d_in[76]; in[76] = (const float*)d_in[77];    // fc1
    for (int i = 0; i < 4; ++i) in[77 + i] = (const float*)d_in[68 + i]; // bn1
    in[81] = (const float*)d_in[78]; in[82] = (const float*)d_in[79];    // fc2
    for (int i = 0; i < 4; ++i) in[83 + i] = (const float*)d_in[72 + i]; // bn2
    in[87] = (const float*)d_in[82]; in[88] = (const float*)d_in[83];    // out
  }
  const float* x = in[0];

  float* ws   = (float*)d_ws;
  float* bufA = ws;                  // 802816 floats (64*112*112 max)
  float* bufB = ws + 802816;
  float* sa   = ws + 1605632;        // 9408
  float* q    = ws + 1615040;
  float* kk   = ws + 1624448;
  float* vv   = ws + 1633856;
  float* vatt = ws + 1643264;
  float* km   = ws + 1652672;        // 3136
  float* vm   = ws + 1655808;
  float* scal = ws + 1658944;        // 8
  float* fx   = ws + 1658952;        // 34496, 16B aligned
  float* h1   = ws + 1693448;        // 4096
  float* h2   = ws + 1697544;        // 4096
  _Float16* w16base = (_Float16*)(ws + 1701648); // 7,632,896 f16 (15.3MB)
  float* outp = (float*)d_out;
  (void)out_size; (void)ws_size; (void)n_in;

  // Conv geometry tables
  static const int convCi[10] = {3, 64, 64, 128, 128, 256, 256, 256, 512, 512};
  static const int convCo[10] = {64, 64, 128, 128, 256, 256, 256, 512, 512, 512};
  static const int convKp[10] = {32, 576, 576, 1152, 1152, 2304, 2304, 2304, 4608, 4608};
  size_t w16off[10];
  {
    size_t o = 0;
    for (int j = 0; j < 10; ++j) { w16off[j] = o; o += (size_t)convCo[j] * convKp[j]; }
  }

  dim3 b256(256);

  // Weight pre-conversion (f32 -> padded f16), once per launch
  for (int j = 0; j < 10; ++j) {
    int total = convCo[j] * convKp[j];
    wcvt_k<<<(total + 255) / 256, b256, 0, stream>>>(
        in[1 + j * 6], w16base + w16off[j], convCo[j], convCi[j] * 9, convKp[j]);
  }

  // Stem pools + q/k/v + shared attention state
  avgpool_k<<<(37632 + 255) / 256, b256, 0, stream>>>(x, bufA, 3, 224, 2);
  avgpool_k<<<(9408 + 255) / 256, b256, 0, stream>>>(x, sa, 3, 224, 4);
  qkv_k<<<37, b256, 0, stream>>>(sa, q, kk, vv, in[61], in[62], in[63], in[64],
                                 in[65], in[66], 9408);
  attn_equal_k<<<dim3(13, 3), b256, 0, stream>>>(q, kk, vv, vatt);
  kmvm_k<<<1, b256, 0, stream>>>(kk, vatt, km, vm, scal);

  auto launch_conv = [&](int j, const float* src, float* dst, int H) {
    int Co = convCo[j];
    dim3 grid((H * H + 31) / 32, (Co + 127) / 128);
    conv3x3_wmma_k<<<grid, b256, 0, stream>>>(
        src, w16base + w16off[j], in[2 + j * 6], in[3 + j * 6], in[4 + j * 6],
        in[5 + j * 6], in[6 + j * 6], dst, convCi[j], Co, H, convKp[j]);
  };

  // Block 1: 112 -> 56
  launch_conv(0, bufA, bufB, 112);
  launch_conv(1, bufB, bufA, 112);
  maxpool2_k<<<(64 * 56 * 56 + 255) / 256, b256, 0, stream>>>(bufA, bufB, 64, 112);
  attn_asym_k<<<(200704 + 255) / 256, b256, 0, stream>>>(bufB, bufA, km, vm, scal,
                                                         in[67], in[68], 200704);
  // Block 2: 56 -> 28
  launch_conv(2, bufA, bufB, 56);
  launch_conv(3, bufB, bufA, 56);
  maxpool2_k<<<(128 * 28 * 28 + 255) / 256, b256, 0, stream>>>(bufA, bufB, 128, 56);
  attn_asym_k<<<(100352 + 255) / 256, b256, 0, stream>>>(bufB, bufA, km, vm, scal,
                                                         in[69], in[70], 100352);
  // Block 3: 28 -> 14
  launch_conv(4, bufA, bufB, 28);
  launch_conv(5, bufB, bufA, 28);
  launch_conv(6, bufA, bufB, 28);
  maxpool2_k<<<(256 * 14 * 14 + 255) / 256, b256, 0, stream>>>(bufB, bufA, 256, 28);
  attn_asym_k<<<(50176 + 255) / 256, b256, 0, stream>>>(bufA, bufB, km, vm, scal,
                                                        in[71], in[72], 50176);
  // Block 4: 14 -> 7
  launch_conv(7, bufB, bufA, 14);
  launch_conv(8, bufA, bufB, 14);
  launch_conv(9, bufB, bufA, 14);
  maxpool2_k<<<(512 * 7 * 7 + 255) / 256, b256, 0, stream>>>(bufA, bufB, 512, 14);
  attn_asym_k<<<(25088 + 255) / 256, b256, 0, stream>>>(bufB, bufA, km, vm, scal,
                                                        in[73], in[74], 25088);

  // Head
  concat_k<<<(34496 + 255) / 256, b256, 0, stream>>>(vatt, bufA, fx, outp + 10);
  gemv_bn_k<<<512, 256, 0, stream>>>(in[75], fx, in[76], in[77], in[78], in[79],
                                     in[80], h1, 34496, 4096, 0);
  gemv_bn_k<<<512, 256, 0, stream>>>(in[81], h1, in[82], in[83], in[84], in[85],
                                     in[86], h2, 4096, 4096, 1);
  out_softmax_k<<<1, 320, 0, stream>>>(in[87], in[88], h2, outp);
}
